// QLSTMEnhanced_65481071402241
// MI455X (gfx1250) — compile-verified, matrix operand order checked by reading, not run on hardware
//
#include <hip/hip_runtime.h>
#include <math.h>

// ---------------------------------------------------------------------------
// Types
// ---------------------------------------------------------------------------
typedef __bf16 bf16_t;
typedef __bf16 v16bf __attribute__((ext_vector_type(16)));
typedef float  v8f   __attribute__((ext_vector_type(8)));

// ---------------------------------------------------------------------------
// Problem constants (match reference)
// ---------------------------------------------------------------------------
constexpr int T_STEPS = 1024;
constexpr int B_SZ    = 128;
constexpr int D_SZ    = 256;
constexpr int H_SZ    = 256;
constexpr int F_SZ    = 128;
constexpr int DH      = D_SZ + H_SZ;      // 512  (gate GEMM K)
constexpr int G4H     = 4 * H_SZ;         // 1024 (gate GEMM N)
constexpr int BH      = B_SZ * H_SZ;      // 32768

// ---------------------------------------------------------------------------
// WMMA fragment swizzle helpers (CDNA5 16x16x32 bf16 layouts, ISA 7.12.2)
// ---------------------------------------------------------------------------
__device__ __forceinline__ int a_swz_idx(int m, int k, int tilesK) {
  int kk   = k & 31;
  int lane = (m & 15) + (((kk >> 3) & 1) << 4);
  int e    = (kk & 7) + (((kk >> 4) & 1) << 3);
  return (((m >> 4) * tilesK + (k >> 5)) * 32 + lane) * 16 + e;
}
__device__ __forceinline__ int b_swz_idx(int k, int n, int tilesN) {
  int kk   = k & 31;
  int lane = (n & 15) + (((kk >> 3) & 1) << 4);
  int e    = (kk & 7) + (((kk >> 4) & 1) << 3);
  return (((k >> 5) * tilesN + (n >> 4)) * 32 + lane) * 16 + e;
}

__device__ __forceinline__ v8f wmma_bf16(v16bf a, v16bf b, v8f c) {
  return __builtin_amdgcn_wmma_f32_16x16x32_bf16(false, a, false, b,
                                                 (short)0, c, false, false);
}

// ---------------------------------------------------------------------------
// Prep kernels: convert f32 operands to bf16 fragment-swizzled layouts
// ---------------------------------------------------------------------------
__global__ void convert_x_kernel(const float* __restrict__ x, bf16_t* __restrict__ x_swz) {
  int idx = blockIdx.x * blockDim.x + threadIdx.x;          // over T*B*D
  int t   = idx / (B_SZ * D_SZ);
  int r   = idx % (B_SZ * D_SZ);
  int b   = r / D_SZ;
  int d   = r % D_SZ;
  x_swz[(size_t)t * (B_SZ * D_SZ) + a_swz_idx(b, d, D_SZ / 32)] = (bf16_t)x[idx];
}

__global__ void convert_wg_kernel(const float* __restrict__ wg, bf16_t* __restrict__ wg_swz) {
  int idx = blockIdx.x * blockDim.x + threadIdx.x;          // over 512*1024
  int k   = idx / G4H;
  int n   = idx % G4H;
  wg_swz[b_swz_idx(k, n, G4H / 16)] = (bf16_t)wg[idx];
}

__global__ void convert_w1_kernel(const float* __restrict__ w1, bf16_t* __restrict__ w1_swz) {
  int idx = blockIdx.x * blockDim.x + threadIdx.x;          // over 4*256*128
  int g   = idx / (H_SZ * F_SZ);
  int r   = idx % (H_SZ * F_SZ);
  int k   = r / F_SZ;                                       // K = H
  int n   = r % F_SZ;                                       // N = F
  w1_swz[g * (H_SZ * F_SZ) + b_swz_idx(k, n, F_SZ / 16)] = (bf16_t)w1[idx];
}

__global__ void convert_w2_kernel(const float* __restrict__ w2, bf16_t* __restrict__ w2_swz) {
  int idx = blockIdx.x * blockDim.x + threadIdx.x;          // over 4*128*256
  int g   = idx / (F_SZ * H_SZ);
  int r   = idx % (F_SZ * H_SZ);
  int k   = r / H_SZ;                                       // K = F
  int n   = r % H_SZ;                                       // N = H
  w2_swz[g * (F_SZ * H_SZ) + b_swz_idx(k, n, H_SZ / 16)] = (bf16_t)w2[idx];
}

__global__ void init_state_kernel(bf16_t* __restrict__ h_swz, float* __restrict__ c) {
  int idx = blockIdx.x * blockDim.x + threadIdx.x;          // over B*H
  h_swz[idx] = (bf16_t)0.0f;
  c[idx]     = 0.0f;
}

// ---------------------------------------------------------------------------
// Per-step gate kernel: one workgroup per gate (grid.x = 4), 256 threads.
// Phase 1: z[g]  = [x_t | h] @ Wg[:, g*256:(g+1)*256] + bg     (K=512)
// Phase 2: h1    = relu(z[g] @ W1[g] + b1[g])                  (K=256)
// Phase 3: z2[g] = h1 @ W2[g] + b2[g]                          (K=128)
// A-fragments preloaded to VGPRs; B-fragments software-pipelined: loads for
// K-step ks+1 are issued before the two WMMAs of step ks (tile pair), so each
// s_wait_loadcnt has two WMMAs of latency cover.
// ---------------------------------------------------------------------------
__global__ void __launch_bounds__(256)
gate_step_kernel(const bf16_t* __restrict__ x_swz_t,   // [B,D] fragments, this step
                 const bf16_t* __restrict__ h_swz,     // [B,H] fragments
                 const bf16_t* __restrict__ wg_swz,    // [512,1024] B-fragments
                 const bf16_t* __restrict__ w1_swz,    // [4][256,128] B-fragments
                 const bf16_t* __restrict__ w2_swz,    // [4][128,256] B-fragments
                 const float*  __restrict__ bg,        // [1024]
                 const float*  __restrict__ b1,        // [4,128]
                 const float*  __restrict__ b2,        // [4,256]
                 float*        __restrict__ z2)        // [4][B,H] f32
{
  const int g    = blockIdx.x;            // gate
  const int wave = threadIdx.x >> 5;      // 0..7
  const int lane = threadIdx.x & 31;

  __shared__ __align__(32) bf16_t z_lds[B_SZ * H_SZ];   // 64 KB, A-swizzled (tilesK=8)
  __shared__ __align__(32) bf16_t h1_lds[B_SZ * F_SZ];  // 32 KB, A-swizzled (tilesK=4)

  const int m_base = wave * 16 + ((lane >> 4) << 3);    // C-frag row base
  const int nn     = lane & 15;                         // C-frag col within tile

  // ---- Phase 1: gate linear, K = 512 (8 x-tiles then 8 h-tiles) -----------
  {
    v16bf afrag[16];
#pragma unroll
    for (int ks = 0; ks < 8; ++ks)
      afrag[ks] = *(const v16bf*)(x_swz_t + (((size_t)wave * 8 + ks) * 32 + lane) * 16);
#pragma unroll
    for (int ks = 0; ks < 8; ++ks)
      afrag[8 + ks] = *(const v16bf*)(h_swz + (((size_t)wave * 8 + ks) * 32 + lane) * 16);

    const bf16_t* wgp = wg_swz + (size_t)lane * 16;     // + (ks*64 + tn)*512
    for (int j = 0; j < 16; j += 2) {                   // N-tile pairs
      const int tn0 = g * 16 + j;
      v8f acc0 = {}, acc1 = {};
      v16bf b0 = *(const v16bf*)(wgp + (size_t)(tn0) * 512);
      v16bf b1 = *(const v16bf*)(wgp + (size_t)(tn0 + 1) * 512);
#pragma unroll
      for (int ks = 0; ks < 16; ++ks) {
        v16bf nb0 = b0, nb1 = b1;
        if (ks < 15) {                                  // prefetch next K-step
          nb0 = *(const v16bf*)(wgp + (size_t)((ks + 1) * 64 + tn0) * 512);
          nb1 = *(const v16bf*)(wgp + (size_t)((ks + 1) * 64 + tn0 + 1) * 512);
        }
        acc0 = wmma_bf16(afrag[ks], b0, acc0);
        acc1 = wmma_bf16(afrag[ks], b1, acc1);
        b0 = nb0; b1 = nb1;
      }
#pragma unroll
      for (int p = 0; p < 2; ++p) {
        const v8f acc = p ? acc1 : acc0;
        const int ncol = (j + p) * 16 + nn;             // 0..255 within gate
        const float bias = bg[g * H_SZ + ncol];
#pragma unroll
        for (int r = 0; r < 8; ++r)
          z_lds[a_swz_idx(m_base + r, ncol, H_SZ / 32)] = (bf16_t)(acc[r] + bias);
      }
    }
  }
  __syncthreads();

  // ---- Phase 2: FFN1 + ReLU, [128,256] x [256,128] -------------------------
  {
    v16bf afrag[8];
#pragma unroll
    for (int ks = 0; ks < 8; ++ks)
      afrag[ks] = *(const v16bf*)(z_lds + ((wave * 8 + ks) * 32 + lane) * 16);

    const bf16_t* w1p = w1_swz + (size_t)g * (H_SZ * F_SZ) + (size_t)lane * 16;
    for (int j = 0; j < 8; j += 2) {                    // N-tile pairs over F
      v8f acc0 = {}, acc1 = {};
      v16bf b0 = *(const v16bf*)(w1p + (size_t)(j) * 512);
      v16bf b1 = *(const v16bf*)(w1p + (size_t)(j + 1) * 512);
#pragma unroll
      for (int ks = 0; ks < 8; ++ks) {
        v16bf nb0 = b0, nb1 = b1;
        if (ks < 7) {
          nb0 = *(const v16bf*)(w1p + (size_t)((ks + 1) * 8 + j) * 512);
          nb1 = *(const v16bf*)(w1p + (size_t)((ks + 1) * 8 + j + 1) * 512);
        }
        acc0 = wmma_bf16(afrag[ks], b0, acc0);
        acc1 = wmma_bf16(afrag[ks], b1, acc1);
        b0 = nb0; b1 = nb1;
      }
#pragma unroll
      for (int p = 0; p < 2; ++p) {
        const v8f acc = p ? acc1 : acc0;
        const int nf = (j + p) * 16 + nn;
        const float bias = b1[g * F_SZ + nf];
#pragma unroll
        for (int r = 0; r < 8; ++r) {
          float v = acc[r] + bias;
          v = v > 0.0f ? v : 0.0f;
          h1_lds[a_swz_idx(m_base + r, nf, F_SZ / 32)] = (bf16_t)v;
        }
      }
    }
  }
  __syncthreads();

  // ---- Phase 3: FFN2, [128,128] x [128,256] -> z2[g] -----------------------
  {
    v16bf afrag[4];
#pragma unroll
    for (int ks = 0; ks < 4; ++ks)
      afrag[ks] = *(const v16bf*)(h1_lds + ((wave * 4 + ks) * 32 + lane) * 16);

    const bf16_t* w2p = w2_swz + (size_t)g * (F_SZ * H_SZ) + (size_t)lane * 16;
    float* zout = z2 + (size_t)g * BH;
    for (int j = 0; j < 16; j += 2) {                   // N-tile pairs over H
      v8f acc0 = {}, acc1 = {};
      v16bf b0 = *(const v16bf*)(w2p + (size_t)(j) * 512);
      v16bf b1 = *(const v16bf*)(w2p + (size_t)(j + 1) * 512);
#pragma unroll
      for (int ks = 0; ks < 4; ++ks) {
        v16bf nb0 = b0, nb1 = b1;
        if (ks < 3) {
          nb0 = *(const v16bf*)(w2p + (size_t)((ks + 1) * 16 + j) * 512);
          nb1 = *(const v16bf*)(w2p + (size_t)((ks + 1) * 16 + j + 1) * 512);
        }
        acc0 = wmma_bf16(afrag[ks], b0, acc0);
        acc1 = wmma_bf16(afrag[ks], b1, acc1);
        b0 = nb0; b1 = nb1;
      }
#pragma unroll
      for (int p = 0; p < 2; ++p) {
        const v8f acc = p ? acc1 : acc0;
        const int nh = (j + p) * 16 + nn;
        const float bias = b2[g * H_SZ + nh];
#pragma unroll
        for (int r = 0; r < 8; ++r)
          zout[(size_t)(m_base + r) * H_SZ + nh] = acc[r] + bias;
      }
    }
  }
}

// ---------------------------------------------------------------------------
// Per-step combine: gating nonlinearities + state update + output write.
// ---------------------------------------------------------------------------
__global__ void __launch_bounds__(256)
combine_kernel(const float* __restrict__ z2,    // [4][B,H]
               float*       __restrict__ c,     // [B,H] state (rw)
               float*       __restrict__ h_out, // d_out + t*B*H
               bf16_t*      __restrict__ h_swz) // [B,H] fragments for next step
{
  const int idx = blockIdx.x * blockDim.x + threadIdx.x;  // 0..BH-1
  const int b   = idx >> 8;         // H=256
  const int col = idx & 255;

  const float zf = z2[0 * BH + idx];
  const float zi = z2[1 * BH + idx];
  const float zg = z2[2 * BH + idx];
  const float zo = z2[3 * BH + idx];

  const float fg = 1.0f / (1.0f + expf(-zf));
  const float ig = 1.0f / (1.0f + expf(-zi));
  const float gg = tanhf(zg);
  const float og = 1.0f / (1.0f + expf(-zo));

  const float cn = fg * c[idx] + ig * gg;
  c[idx] = cn;
  const float h = og * tanhf(cn);
  h_out[idx] = h;
  h_swz[a_swz_idx(b, col, H_SZ / 32)] = (bf16_t)h;
}

// ---------------------------------------------------------------------------
// Host launch
// ---------------------------------------------------------------------------
extern "C" void kernel_launch(void* const* d_in, const int* in_sizes, int n_in,
                              void* d_out, int out_size, void* d_ws, size_t ws_size,
                              hipStream_t stream) {
  (void)in_sizes; (void)n_in; (void)out_size; (void)ws_size;

  const float* x  = (const float*)d_in[0];   // [T,B,D]
  const float* Wg = (const float*)d_in[1];   // [512,1024]
  const float* bg = (const float*)d_in[2];   // [1024]
  const float* W1 = (const float*)d_in[3];   // [4,256,128]
  const float* b1 = (const float*)d_in[4];   // [4,128]
  const float* W2 = (const float*)d_in[5];   // [4,128,256]
  const float* b2 = (const float*)d_in[6];   // [4,256]
  float* out = (float*)d_out;                // [T,B,H] + hx[B,H] + cx[B,H]

  char* ws = (char*)d_ws;
  size_t off = 0;
  auto carve = [&](size_t bytes) { char* p = ws + off; off = (off + bytes + 255) & ~(size_t)255; return p; };
  bf16_t* x_swz  = (bf16_t*)carve((size_t)T_STEPS * B_SZ * D_SZ * sizeof(bf16_t)); // 64 MB
  bf16_t* wg_swz = (bf16_t*)carve((size_t)DH * G4H * sizeof(bf16_t));              //  1 MB
  bf16_t* w1_swz = (bf16_t*)carve((size_t)4 * H_SZ * F_SZ * sizeof(bf16_t));
  bf16_t* w2_swz = (bf16_t*)carve((size_t)4 * F_SZ * H_SZ * sizeof(bf16_t));
  bf16_t* h_swz  = (bf16_t*)carve((size_t)BH * sizeof(bf16_t));
  float*  c_st   = (float*) carve((size_t)BH * sizeof(float));
  float*  z2     = (float*) carve((size_t)4 * BH * sizeof(float));

  convert_x_kernel <<<(T_STEPS * B_SZ * D_SZ) / 256, 256, 0, stream>>>(x, x_swz);
  convert_wg_kernel<<<(DH * G4H) / 256,              256, 0, stream>>>(Wg, wg_swz);
  convert_w1_kernel<<<(4 * H_SZ * F_SZ) / 256,       256, 0, stream>>>(W1, w1_swz);
  convert_w2_kernel<<<(4 * F_SZ * H_SZ) / 256,       256, 0, stream>>>(W2, w2_swz);
  init_state_kernel<<<BH / 256,                      256, 0, stream>>>(h_swz, c_st);

  for (int t = 0; t < T_STEPS; ++t) {
    gate_step_kernel<<<4, 256, 0, stream>>>(
        x_swz + (size_t)t * B_SZ * D_SZ, h_swz, wg_swz, w1_swz, w2_swz,
        bg, b1, b2, z2);
    combine_kernel<<<BH / 256, 256, 0, stream>>>(
        z2, c_st, out + (size_t)t * BH, h_swz);
  }

  hipMemcpyAsync(out + (size_t)T_STEPS * BH, out + (size_t)(T_STEPS - 1) * BH,
                 (size_t)BH * sizeof(float), hipMemcpyDeviceToDevice, stream);
  hipMemcpyAsync(out + (size_t)T_STEPS * BH + BH, c_st,
                 (size_t)BH * sizeof(float), hipMemcpyDeviceToDevice, stream);
}
